// MambaFusion_52003464020734
// MI455X (gfx1250) — compile-verified, matrix operand order checked by reading, not run on hardware
//
#include <hip/hip_runtime.h>
#include <hip/hip_bf16.h>

#define BZc 16
#define SEQc 5
#define C_DIM 384
#define DSTATEc 16
#define DCONVc 4
#define DIc 768
#define DTRc 24
#define NLAYERc 4
#define NTOKc 960
#define T_DIM 962
#define NROWS (BZc * T_DIM)          /* 15392, multiple of 16 */

typedef __bf16 bf16_t;
typedef __attribute__((ext_vector_type(16))) __bf16 v16bf;
typedef __attribute__((ext_vector_type(8)))  __bf16 v8bf;
typedef __attribute__((ext_vector_type(8)))  float  v8f;

static __device__ __forceinline__ float sigmoidf_(float x) { return 1.0f / (1.0f + __expf(-x)); }
static __device__ __forceinline__ float siluf_(float x)    { return x * sigmoidf_(x); }
static __device__ __forceinline__ float softplusf_(float x){ return (x > 20.0f) ? x : log1pf(__expf(x)); }

#define CEILDIV(a, b) (((a) + (b) - 1) / (b))

static __device__ __forceinline__ v16bf ldfrag(const bf16_t* __restrict__ row, int koff)
{
  // ISA 16-bit fragment layout: element i   -> K = kb + i       (i = 0..7)
  //                             element 8+i -> K = kb + 16 + i
  v8bf lo = *(const v8bf*)(row + koff);
  v8bf hi = *(const v8bf*)(row + koff + 16);
  v16bf f;
#pragma unroll
  for (int i = 0; i < 8; ++i) { f[i] = lo[i]; f[8 + i] = hi[i]; }
  return f;
}

// ---------------------------------------------------------------------------
// Generic WMMA GEMM: Out[m,n] = sum_k A[m,k]*W[n,k] (+bias[n])
// A: MxK bf16 row-major, W: NxK bf16 row-major. M%16==0, N%64==0, K%32==0.
// One 16x64 output strip per wave (4 accumulators reuse one A fragment),
// 8 waves per 256-thread block.
// ---------------------------------------------------------------------------
__global__ __launch_bounds__(256)
void gemm_wmma_bf16(const bf16_t* __restrict__ A, const bf16_t* __restrict__ W,
                    const float* __restrict__ bias,
                    float* __restrict__ outF, bf16_t* __restrict__ outB,
                    int M, int N, int K)
{
  const int lane = threadIdx.x & 31;
  const int wave = threadIdx.x >> 5;
  const int tm = blockIdx.x;
  const int tn0 = (blockIdx.y * 8 + wave) * 4;   // first of 4 consecutive N-tiles
  if (tn0 * 16 >= N) return;                     // wave-uniform exit, EXEC stays full

  const int hsel = lane >> 4;                    // 0: lanes 0-15, 1: lanes 16-31
  const int kb   = hsel * 8;
  const bf16_t* __restrict__ arow = A + (size_t)(tm * 16 + (lane & 15)) * K;
  const bf16_t* __restrict__ brow = W + (size_t)(tn0 * 16 + (lane & 15)) * K;
  const size_t bstep = (size_t)16 * K;           // stride between B n-tiles

  v8f acc0 = {0.f,0.f,0.f,0.f,0.f,0.f,0.f,0.f};
  v8f acc1 = acc0, acc2 = acc0, acc3 = acc0;

  for (int k0 = 0; k0 < K; k0 += 32) {
    const int ko = k0 + kb;
    v16bf a  = ldfrag(arow, ko);
    v16bf b0 = ldfrag(brow,             ko);
    v16bf b1 = ldfrag(brow + bstep,     ko);
    v16bf b2 = ldfrag(brow + 2 * bstep, ko);
    v16bf b3 = ldfrag(brow + 3 * bstep, ko);
    acc0 = __builtin_amdgcn_wmma_f32_16x16x32_bf16(false, a, false, b0, (short)0, acc0, false, false);
    acc1 = __builtin_amdgcn_wmma_f32_16x16x32_bf16(false, a, false, b1, (short)0, acc1, false, false);
    acc2 = __builtin_amdgcn_wmma_f32_16x16x32_bf16(false, a, false, b2, (short)0, acc2, false, false);
    acc3 = __builtin_amdgcn_wmma_f32_16x16x32_bf16(false, a, false, b3, (short)0, acc3, false, false);
  }

  // C/D layout: VGPR r -> row = r + 8*hsel, col = lane&15
  const int rowBase = tm * 16 + hsel * 8;
  v8f accs[4] = {acc0, acc1, acc2, acc3};
#pragma unroll
  for (int j = 0; j < 4; ++j) {
    const int col = (tn0 + j) * 16 + (lane & 15);
    const float bv = bias ? bias[col] : 0.0f;
#pragma unroll
    for (int r = 0; r < 8; ++r) {
      size_t o = (size_t)(rowBase + r) * N + col;
      float v = accs[j][r] + bv;
      if (outF) outF[o] = v;
      if (outB) outB[o] = (bf16_t)v;
    }
  }
}

// ---------------------------------------------------------------------------
// fp32 -> bf16 convert (flat) and batched 2D zero-padding convert
// ---------------------------------------------------------------------------
__global__ void cvt_k(const float* __restrict__ s, bf16_t* __restrict__ d, int count)
{
  int i = blockIdx.x * 256 + threadIdx.x;
  if (i < count) d[i] = (bf16_t)s[i];
}

__global__ void cvt_pad_k(const float* __restrict__ s, bf16_t* __restrict__ d,
                          int batch, int dR, int dC, int sR, int sC)
{
  int i = blockIdx.x * 256 + threadIdx.x;
  int total = batch * dR * dC;
  if (i >= total) return;
  int c  = i % dC;
  int r  = (i / dC) % dR;
  int bt = i / (dC * dR);
  float v = (r < sR && c < sC) ? s[((size_t)bt * sR + r) * sC + c] : 0.0f;
  d[i] = (bf16_t)v;
}

// ---------------------------------------------------------------------------
// Token assembly: channel-shuffled img/lid/rad frames + gps + pos_emb -> x
// ---------------------------------------------------------------------------
__global__ void assemble_k(const float* __restrict__ img, const float* __restrict__ lid,
                           const float* __restrict__ rad, const float* __restrict__ gps,
                           const float* __restrict__ pos, float* __restrict__ x)
{
  int idx = blockIdx.x * 256 + threadIdx.x;
  if (idx >= NROWS * C_DIM) return;
  int c = idx % C_DIM;
  int n = idx / C_DIM;
  int t = n % T_DIM;
  int b = n / T_DIM;
  float v;
  if (t < NTOKc) {
    int F = t >> 6;           // frame 0..14
    int ij = t & 63;          // i*8+j
    int f = F % SEQc;
    int grp = F / SEQc;       // 0: img-shuffle, 1: lid-shuffle, 2: rad-shuffle
    int band = (c < 128) ? 0 : (c < 256 ? 1 : 2);
    // sel table: grp0:{img,lid,rad} grp1:{lid,rad,img} grp2:{rad,img,lid}
    int sel = (grp + band) % 3;
    const float* src = (sel == 0) ? img : (sel == 1) ? lid : rad;
    v = src[(((size_t)(b * SEQc + f) * C_DIM + c) << 6) + ij];
  } else {
    v = gps[(size_t)(b * 2 + (t - NTOKc)) * C_DIM + c];
  }
  x[idx] = v + pos[(size_t)t * C_DIM + c];
}

// ---------------------------------------------------------------------------
// LayerNorm over C=384; one wave per row; optional f32 / bf16 outputs
// ---------------------------------------------------------------------------
__global__ __launch_bounds__(256)
void layernorm_k(const float* __restrict__ x, const float* __restrict__ g,
                 const float* __restrict__ bta, float* __restrict__ outF,
                 bf16_t* __restrict__ outB, int nrows)
{
  int lane = threadIdx.x & 31;
  int wave = threadIdx.x >> 5;
  int row = blockIdx.x * 8 + wave;
  if (row >= nrows) return;
  const float* xr = x + (size_t)row * C_DIM;
  float s = 0.f;
  for (int c = lane; c < C_DIM; c += 32) s += xr[c];
  for (int off = 16; off; off >>= 1) s += __shfl_xor(s, off);
  float mean = s * (1.0f / C_DIM);
  float var = 0.f;
  for (int c = lane; c < C_DIM; c += 32) { float d = xr[c] - mean; var += d * d; }
  for (int off = 16; off; off >>= 1) var += __shfl_xor(var, off);
  float rstd = rsqrtf(var * (1.0f / C_DIM) + 1e-5f);
  for (int c = lane; c < C_DIM; c += 32) {
    float val = (xr[c] - mean) * rstd * g[c] + bta[c];
    size_t o = (size_t)row * C_DIM + c;
    if (outF) outF[o] = val;
    if (outB) outB[o] = (bf16_t)val;
  }
}

// ---------------------------------------------------------------------------
// Depthwise causal conv (DCONV=4) + bias + SiLU.
// xz is in ORIGINAL time order; output xh is in VIRTUAL (scan) order:
// dir==0 -> same order; dir==1 -> virtual t reads original T-1-t.
// ---------------------------------------------------------------------------
__global__ void conv_silu_k(const bf16_t* __restrict__ xz, const float* __restrict__ cw,
                            const float* __restrict__ cb, bf16_t* __restrict__ xhB, int dir)
{
  int idx = blockIdx.x * 256 + threadIdx.x;
  if (idx >= NROWS * DIc) return;
  int d = idx % DIc;
  int bt = idx / DIc;
  int tv = bt % T_DIM;
  int b = bt / T_DIM;
  float acc = cb[d];
#pragma unroll
  for (int j = 0; j < DCONVc; ++j) {
    int tvj = tv - (DCONVc - 1) + j;
    if (tvj < 0) continue;
    int to = dir ? (T_DIM - 1 - tvj) : tvj;
    acc += cw[d * DCONVc + j] * (float)xz[(size_t)(b * T_DIM + to) * (2 * DIc) + d];
  }
  xhB[(size_t)bt * DIc + d] = (bf16_t)siluf_(acc);
}

// dt columns of xdbl (N x 64 padded) -> zero-padded bf16 (N x 32) for WMMA K=32
__global__ void dtpad_k(const float* __restrict__ xdbl, bf16_t* __restrict__ dtp)
{
  int idx = blockIdx.x * 256 + threadIdx.x;
  if (idx >= NROWS * 32) return;
  int r = idx & 31;
  int n = idx >> 5;
  dtp[idx] = (bf16_t)((r < DTRc) ? xdbl[(size_t)n * 64 + r] : 0.0f);
}

// ---------------------------------------------------------------------------
// Selective scan. 16 lanes per (b,d) channel: one lane per state s.
// h = h*exp(dt*A_s) + dt*x*B_s ; y = sum_s h*C_s + D*x ; y *= silu(z)
// Everything (dtraw/xh/xdbl/y) is in virtual order; z read from xz at original
// index. Intra-16-lane shfl_xor reduction for the state sum.
// ---------------------------------------------------------------------------
__global__ __launch_bounds__(256)
void scan_k(const float* __restrict__ dtraw, const float* __restrict__ dtb,
            const bf16_t* __restrict__ xhB, const float* __restrict__ xdbl,
            const bf16_t* __restrict__ xz, const float* __restrict__ a_log,
            const float* __restrict__ Dp, bf16_t* __restrict__ yB, int dir)
{
  int tid = blockIdx.x * 256 + threadIdx.x;
  if (tid >= BZc * DIc * DSTATEc) return;
  int s = tid & 15;
  int ch = tid >> 4;                 // b*DI + d
  int d = ch % DIc;
  int b = ch / DIc;
  const float A = -__expf(a_log[(size_t)d * DSTATEc + s]);
  const float bdt = dtb[d];
  const float dpar = Dp[d];
  float h = 0.0f;
  for (int tv = 0; tv < T_DIM; ++tv) {
    size_t n = (size_t)b * T_DIM + tv;
    float dt = softplusf_(dtraw[n * DIc + d] + bdt);
    float xv = (float)xhB[n * DIc + d];
    const float* bcrow = xdbl + n * 64;
    float Bv = bcrow[24 + s];
    float Cv = bcrow[40 + s];
    h = h * __expf(dt * A) + dt * xv * Bv;
    float p = h * Cv;
    p += __shfl_xor(p, 1);
    p += __shfl_xor(p, 2);
    p += __shfl_xor(p, 4);
    p += __shfl_xor(p, 8);
    if (s == 0) {
      int to = dir ? (T_DIM - 1 - tv) : tv;
      float z = (float)xz[(size_t)(b * T_DIM + to) * (2 * DIc) + DIc + d];
      yB[n * DIc + d] = (bf16_t)((p + dpar * xv) * siluf_(z));
    }
  }
}

// x_new = x_bm * (leaky_relu(fc2(x_flip)) + x_fm); fc2 was computed on
// original order, so its flipped-frame value lives at row (b, T-1-t).
__global__ void combine_k(const float* __restrict__ yF, const float* __restrict__ yBk,
                          const float* __restrict__ fc2o, float* __restrict__ x)
{
  int idx = blockIdx.x * 256 + threadIdx.x;
  if (idx >= NROWS * C_DIM) return;
  int c = idx % C_DIM;
  int n = idx / C_DIM;
  int t = n % T_DIM;
  int b = n / T_DIM;
  float f2 = fc2o[((size_t)(b * T_DIM + (T_DIM - 1 - t))) * C_DIM + c];
  float relu = (f2 >= 0.0f) ? f2 : 0.2f * f2;
  x[idx] = yBk[idx] * (relu + yF[idx]);
}

// Scatter final LN result into (img_out, lid_out, rad_out, pos_out)
__global__ void output_k(const float* __restrict__ xf, float* __restrict__ out)
{
  int idx = blockIdx.x * 256 + threadIdx.x;
  if (idx >= NROWS * C_DIM) return;
  int c = idx % C_DIM;
  int n = idx / C_DIM;
  int t = n % T_DIM;
  int b = n / T_DIM;
  float v = xf[idx];
  const size_t BLK = (size_t)BZc * SEQc * C_DIM * 64;   // 1,966,080
  if (t < NTOKc) {
    int F = t >> 6;
    int ij = t & 63;
    int f = F % SEQc;
    int grp = F / SEQc;
    size_t o = (((size_t)(b * SEQc + f) * C_DIM + c) << 6) + ij;
    out[(size_t)grp * BLK + o] = v;
  } else {
    out[3 * BLK + (size_t)(b * 2 + (t - NTOKc)) * C_DIM + c] = v;
  }
}

// ---------------------------------------------------------------------------
extern "C" void kernel_launch(void* const* d_in, const int* in_sizes, int n_in,
                              void* d_out, int out_size, void* d_ws, size_t ws_size,
                              hipStream_t stream)
{
  (void)in_sizes; (void)n_in; (void)out_size; (void)ws_size;
  const float* img      = (const float*)d_in[0];
  const float* lid      = (const float*)d_in[1];
  const float* rad      = (const float*)d_in[2];
  const float* gps      = (const float*)d_in[3];
  const float* pos_emb  = (const float*)d_in[4];
  const float* ln1_g    = (const float*)d_in[5];
  const float* ln1_b    = (const float*)d_in[6];
  const float* fc1_w    = (const float*)d_in[7];
  const float* fc1_b    = (const float*)d_in[8];
  const float* fc2_w    = (const float*)d_in[9];
  const float* fc2_b    = (const float*)d_in[10];
  const float* in_proj  = (const float*)d_in[11];
  const float* conv_w   = (const float*)d_in[12];
  const float* conv_b   = (const float*)d_in[13];
  const float* x_proj   = (const float*)d_in[14];
  const float* dt_projw = (const float*)d_in[15];
  const float* dt_projb = (const float*)d_in[16];
  const float* A_log    = (const float*)d_in[17];
  const float* D_param  = (const float*)d_in[18];
  const float* out_proj = (const float*)d_in[19];
  const float* lnf_g    = (const float*)d_in[20];
  const float* lnf_b    = (const float*)d_in[21];
  float* out = (float*)d_out;

  // ---- carve workspace ----
  char* ws = (char*)d_ws;
  size_t off = 0;
  auto take = [&](size_t bytes) -> char* {
    char* p = ws + off;
    off += (bytes + 255) & ~(size_t)255;
    return p;
  };
  const size_t NC = (size_t)NROWS * C_DIM;
  float*  xbuf  = (float*)  take(NC * 4);
  bf16_t* xlnB  = (bf16_t*) take(NC * 2);
  bf16_t* fc1B  = (bf16_t*) take(NC * 2);
  float*  fc2o  = (float*)  take(NC * 4);
  float*  yFo   = (float*)  take(NC * 4);
  float*  yBo   = (float*)  take(NC * 4);
  bf16_t* xzB   = (bf16_t*) take((size_t)NROWS * 2 * DIc * 2);
  bf16_t* xhB   = (bf16_t*) take((size_t)NROWS * DIc * 2);
  float*  xdbl  = (float*)  take((size_t)NROWS * 64 * 4);
  bf16_t* dtp   = (bf16_t*) take((size_t)NROWS * 32 * 2);
  float*  dtraw = (float*)  take((size_t)NROWS * DIc * 4);
  bf16_t* yB    = (bf16_t*) take((size_t)NROWS * DIc * 2);
  bf16_t* wfc1  = (bf16_t*) take((size_t)NLAYERc * C_DIM * C_DIM * 2);
  bf16_t* wfc2  = (bf16_t*) take((size_t)NLAYERc * C_DIM * C_DIM * 2);
  bf16_t* winp  = (bf16_t*) take((size_t)NLAYERc * 2 * 2 * DIc * C_DIM * 2);
  bf16_t* wxp   = (bf16_t*) take((size_t)NLAYERc * 2 * 64 * DIc * 2);
  bf16_t* wdt   = (bf16_t*) take((size_t)NLAYERc * 2 * DIc * 32 * 2);
  bf16_t* wout  = (bf16_t*) take((size_t)NLAYERc * 2 * C_DIM * DIc * 2);

  // ---- weight conversion to bf16 (once per call) ----
  {
    int n1 = NLAYERc * C_DIM * C_DIM;
    cvt_k<<<CEILDIV(n1, 256), 256, 0, stream>>>(fc1_w, wfc1, n1);
    cvt_k<<<CEILDIV(n1, 256), 256, 0, stream>>>(fc2_w, wfc2, n1);
    int n2 = NLAYERc * 2 * 2 * DIc * C_DIM;
    cvt_k<<<CEILDIV(n2, 256), 256, 0, stream>>>(in_proj, winp, n2);
    int n3 = NLAYERc * 2 * C_DIM * DIc;
    cvt_k<<<CEILDIV(n3, 256), 256, 0, stream>>>(out_proj, wout, n3);
    int n4 = NLAYERc * 2 * 64 * DIc;
    cvt_pad_k<<<CEILDIV(n4, 256), 256, 0, stream>>>(x_proj, wxp, NLAYERc * 2, 64, DIc,
                                                    DTRc + 2 * DSTATEc, DIc);
    int n5 = NLAYERc * 2 * DIc * 32;
    cvt_pad_k<<<CEILDIV(n5, 256), 256, 0, stream>>>(dt_projw, wdt, NLAYERc * 2, DIc, 32,
                                                    DIc, DTRc);
  }

  // ---- token assembly ----
  {
    int n = NROWS * C_DIM;
    assemble_k<<<CEILDIV(n, 256), 256, 0, stream>>>(img, lid, rad, gps, pos_emb, xbuf);
  }

  auto gemm = [&](const bf16_t* A, const bf16_t* W, const float* bias,
                  float* oF, bf16_t* oB, int Ncols, int K) {
    dim3 grid(NROWS / 16, CEILDIV(Ncols / 64, 8));   // 4 N-tiles per wave
    gemm_wmma_bf16<<<grid, 256, 0, stream>>>(A, W, bias, oF, oB, NROWS, Ncols, K);
  };

  const int nElemC  = NROWS * C_DIM;
  const int nElemDI = NROWS * DIc;

  for (int l = 0; l < NLAYERc; ++l) {
    // LayerNorm -> bf16
    layernorm_k<<<CEILDIV(NROWS, 8), 256, 0, stream>>>(
        xbuf, ln1_g + l * C_DIM, ln1_b + l * C_DIM, (float*)nullptr, xlnB, NROWS);
    // fc1 (+bias) -> bf16
    gemm(xlnB, wfc1 + (size_t)l * C_DIM * C_DIM, fc1_b + l * C_DIM,
         nullptr, fc1B, C_DIM, C_DIM);
    // fc2 (+bias) on same rows (flip handled at combine)
    gemm(fc1B, wfc2 + (size_t)l * C_DIM * C_DIM, fc2_b + l * C_DIM,
         fc2o, nullptr, C_DIM, C_DIM);

    for (int dir = 0; dir < 2; ++dir) {
      int ld = l * 2 + dir;
      // in_proj: (N,384) x (1536,384)^T -> xz bf16 (original order)
      gemm(fc1B, winp + (size_t)ld * 2 * DIc * C_DIM, nullptr,
           nullptr, xzB, 2 * DIc, C_DIM);
      // causal depthwise conv + SiLU, into virtual (scan) order
      conv_silu_k<<<CEILDIV(nElemDI, 256), 256, 0, stream>>>(
          xzB, conv_w + (size_t)ld * DIc * DCONVc, conv_b + (size_t)ld * DIc, xhB, dir);
      // x_proj: (N,768) x (64,768)^T -> xdbl fp32 (padded N=64)
      gemm(xhB, wxp + (size_t)ld * 64 * DIc, nullptr, xdbl, nullptr, 64, DIc);
      // pad dt part to K=32 bf16
      dtpad_k<<<CEILDIV(NROWS * 32, 256), 256, 0, stream>>>(xdbl, dtp);
      // dt_proj: (N,32) x (768,32)^T -> dtraw fp32
      gemm(dtp, wdt + (size_t)ld * DIc * 32, nullptr, dtraw, nullptr, DIc, 32);
      // selective scan + D-skip + SiLU(z) gate -> y bf16
      scan_k<<<CEILDIV(BZc * DIc * DSTATEc, 256), 256, 0, stream>>>(
          dtraw, dt_projb + (size_t)ld * DIc, xhB, xdbl, xzB,
          A_log + (size_t)ld * DIc * DSTATEc, D_param + (size_t)ld * DIc, yB, dir);
      // out_proj: (N,768) x (384,768)^T -> per-direction fp32
      gemm(yB, wout + (size_t)ld * C_DIM * DIc, nullptr,
           dir ? yBo : yFo, nullptr, C_DIM, DIc);
    }
    // x = x_bm * (leaky(fc2(x_flip)) + x_fm)
    combine_k<<<CEILDIV(nElemC, 256), 256, 0, stream>>>(yFo, yBo, fc2o, xbuf);
  }

  // final LayerNorm (fp32 out, reuse yFo) then scatter to outputs
  layernorm_k<<<CEILDIV(NROWS, 8), 256, 0, stream>>>(
      xbuf, lnf_g, lnf_b, yFo, (bf16_t*)nullptr, NROWS);
  output_k<<<CEILDIV(nElemC, 256), 256, 0, stream>>>(yFo, out);
}
